// PositionSensitiveNetwork_40286793237083
// MI455X (gfx1250) — matrix operand order repaired and checked by reading, not compiled, not run on hardware
//
#include <hip/hip_runtime.h>
#include <hip/hip_bf16.h>

typedef __bf16 v16bf __attribute__((ext_vector_type(16)));
typedef __bf16 v8bf  __attribute__((ext_vector_type(8)));
typedef float  v8f   __attribute__((ext_vector_type(8)));

#if defined(__AMDGCN__) && __has_builtin(__builtin_amdgcn_global_load_async_to_lds_b128) && __has_builtin(__builtin_amdgcn_s_wait_asynccnt)
#define USE_ASYNC 1
#else
#define USE_ASYNC 0
#endif

// Problem constants
static constexpr int FH   = 38;      // feature H
static constexpr int FW   = 50;      // feature W
static constexpr int HWv  = 1900;    // valid pixels
static constexpr int NP   = 1920;    // padded pixels (multiple of 128)
static constexpr int CIN  = 512;
static constexpr int CMID = 1024;
static constexpr int CCLS = 1029;    // 7*7*21
static constexpr int CBOX = 4116;    // 7*7*21*4
static constexpr int MP_D = 1024;    // padded rows, w_double
static constexpr int MP_S = 1152;    // padded rows, w_score (ceil(1029/128)*128)
static constexpr int MP_B = 4224;    // padded rows, w_bbox  (ceil(4116/128)*128)
static constexpr int KROI = 300;

#if USE_ASYNC
// Builtin signature (from hipcc diagnostic): param 0 is
// 'int __attribute__((vector_size(16))) __device__ *' i.e. v4i in AS(1);
// param 1 is the LDS destination (AS(3)); then imm offset, imm cpol.
typedef int v4i_vs __attribute__((__vector_size__(16)));
typedef __attribute__((address_space(1))) v4i_vs* gas_b128;
typedef __attribute__((address_space(3))) v4i_vs* las_b128;

__device__ __forceinline__ void async_copy16(const __bf16* g, __bf16* l) {
  __builtin_amdgcn_global_load_async_to_lds_b128((gas_b128)g, (las_b128)l, 0, 0);
}
#endif

// ---------------- conversion kernels ----------------
__global__ void cvt_w_bf16(const float* __restrict__ in, __bf16* __restrict__ out,
                           int n_valid, int n_total) {
  int idx = blockIdx.x * blockDim.x + threadIdx.x;
  if (idx >= n_total) return;
  float v = (idx < n_valid) ? in[idx] : 0.f;
  out[idx] = (__bf16)v;
}

// feature [C=512][1900] fp32 -> featT [1920][512] bf16 (pixel-major, channel contiguous)
__global__ void cvt_featT(const float* __restrict__ feat, __bf16* __restrict__ featT) {
  int idx = blockIdx.x * blockDim.x + threadIdx.x;   // NP*CIN total
  if (idx >= NP * CIN) return;
  int c = idx & (CIN - 1);
  int n = idx >> 9;                                  // CIN = 512 = 2^9
  float v = (n < HWv) ? feat[(size_t)c * HWv + n] : 0.f;
  featT[idx] = (__bf16)v;
}

// ---------------- WMMA GEMM ----------------
// C[M x NP] = relu(A[Mpad x K] * BT[NP x K]^T + bias)
// Block tile 128x128, 8 waves, wave tile 64x32 (4x2 of 16x16x32 WMMA), K-panel 64.
// MODE 0: out = bf16, written transposed as out[n*ldOut + m]  (feat2T)
// MODE 1: out = fp32, out[m*ldOut + n], rows guarded by Mvalid (maps)
template<int MODE>
__global__ __launch_bounds__(256) void gemm_wmma(
    const __bf16* __restrict__ A,    // [Mpad][K] row-major bf16
    const __bf16* __restrict__ BT,   // [NP][K]   row-major bf16 (pixel-major)
    const float*  __restrict__ bias, // [Mvalid]
    void* __restrict__ outp,
    int Mvalid, int Kdim, int ldOut)
{
  __shared__ alignas(16) __bf16 As[2][128 * 64];
  __shared__ alignas(16) __bf16 Bs[2][128 * 64];

  const int tid  = threadIdx.x;
  const int lane = tid & 31;            // wave32
  const int wv   = tid >> 5;            // 8 waves
  const int l16  = lane & 15;
  const int hf   = lane >> 4;
  const int wm   = (wv & 1) * 64;       // 2 waves along M
  const int wn   = (wv >> 1) * 32;      // 4 waves along N
  const int mBlock = blockIdx.y * 128;
  const int nBlock = blockIdx.x * 128;

  v8f acc[4][2];
  #pragma unroll
  for (int i = 0; i < 4; ++i)
    #pragma unroll
    for (int j = 0; j < 2; ++j) {
      v8f z = {0.f, 0.f, 0.f, 0.f, 0.f, 0.f, 0.f, 0.f};
      acc[i][j] = z;
    }

  const int T = Kdim / 64;   // K-panels

  auto compute = [&](int b) {
    #pragma unroll
    for (int kk = 0; kk < 64; kk += 32) {
      union FragU { struct { float4 lo, hi; } p; v16bf v; } fa[4], fb[2];
      // A frag (16x32): lane = row m, K chunks [kk+8h,+8) and [kk+16+8h,+8)
      #pragma unroll
      for (int ms = 0; ms < 4; ++ms) {
        int row = wm + ms * 16 + l16;
        fa[ms].p.lo = *reinterpret_cast<const float4*>(&As[b][row * 64 + kk + 8 * hf]);
        fa[ms].p.hi = *reinterpret_cast<const float4*>(&As[b][row * 64 + kk + 16 + 8 * hf]);
      }
      // B frag (32x16): lane = col n, K chunks [kk+8h,+8) and [kk+16+8h,+8)
      #pragma unroll
      for (int ns = 0; ns < 2; ++ns) {
        int row = wn + ns * 16 + l16;
        fb[ns].p.lo = *reinterpret_cast<const float4*>(&Bs[b][row * 64 + kk + 8 * hf]);
        fb[ns].p.hi = *reinterpret_cast<const float4*>(&Bs[b][row * 64 + kk + 16 + 8 * hf]);
      }
      #pragma unroll
      for (int ms = 0; ms < 4; ++ms)
        #pragma unroll
        for (int ns = 0; ns < 2; ++ns)
          acc[ms][ns] = __builtin_amdgcn_wmma_f32_16x16x32_bf16(
              false, fa[ms].v, false, fb[ns].v, (short)0, acc[ms][ns], false, false);
    }
  };

#if USE_ASYNC
  // Double-buffered: async copies go straight to LDS (ASYNCcnt-tracked);
  // wait-asynccnt + barrier at the top of each iter protects buffer b, and the
  // same barrier proves all waves finished reading buffer b^1 before reissue.
  auto issue = [&](int buf, int t) {
    int k0 = t * 64;
    #pragma unroll
    for (int i = 0; i < 4; ++i) {
      int idx = tid + i * 256;
      int row = idx >> 3;
      int col = (idx & 7) * 8;
      async_copy16(A  + (size_t)(mBlock + row) * Kdim + k0 + col, &As[buf][row * 64 + col]);
      async_copy16(BT + (size_t)(nBlock + row) * Kdim + k0 + col, &Bs[buf][row * 64 + col]);
    }
  };
  issue(0, 0);
  for (int t = 0; t < T; ++t) {
    int b = t & 1;
    __builtin_amdgcn_s_wait_asynccnt(0);
    __syncthreads();
    if (t + 1 < T) issue(b ^ 1, t + 1);
    compute(b);
  }
#else
  // Fallback: register-staged LDS, single buffer.
  for (int t = 0; t < T; ++t) {
    int k0 = t * 64;
    #pragma unroll
    for (int i = 0; i < 4; ++i) {
      int idx = tid + i * 256;
      int row = idx >> 3;
      int col = (idx & 7) * 8;
      *reinterpret_cast<float4*>(&As[0][row * 64 + col]) =
          *reinterpret_cast<const float4*>(A + (size_t)(mBlock + row) * Kdim + k0 + col);
      *reinterpret_cast<float4*>(&Bs[0][row * 64 + col]) =
          *reinterpret_cast<const float4*>(BT + (size_t)(nBlock + row) * Kdim + k0 + col);
    }
    __syncthreads();
    compute(0);
    __syncthreads();
  }
#endif

  // Epilogue: C/D layout -> lane holds column n=l16, rows m = 8*hf + r (contiguous)
  #pragma unroll
  for (int ms = 0; ms < 4; ++ms) {
    int m0 = mBlock + wm + ms * 16 + 8 * hf;
    #pragma unroll
    for (int ns = 0; ns < 2; ++ns) {
      int n = nBlock + wn + ns * 16 + l16;
      if (MODE == 0) {
        __bf16* outT = (__bf16*)outp;
        union { v8bf v; float4 f; } o;
        #pragma unroll
        for (int r = 0; r < 8; ++r) {
          float v = acc[ms][ns][r] + bias[m0 + r];
          o.v[r] = (__bf16)fmaxf(v, 0.f);
        }
        *reinterpret_cast<float4*>(&outT[(size_t)n * ldOut + m0]) = o.f;
      } else {
        float* outF = (float*)outp;
        #pragma unroll
        for (int r = 0; r < 8; ++r) {
          int m = m0 + r;
          if (m < Mvalid) {
            float v = acc[ms][ns][r] + bias[m];
            outF[(size_t)m * ldOut + n] = fmaxf(v, 0.f);
          }
        }
      }
    }
  }
}

// ---------------- PS-ROI pool + bin mean ----------------
__global__ __launch_bounds__(256) void psroi_kernel(
    const float* __restrict__ rois,    // [K][4] = ymin,xmin,ymax,xmax
    const float* __restrict__ scoreM,  // [1029][NP]
    const float* __restrict__ bboxM,   // [4116][NP]
    float* __restrict__ out)           // [300*21] ++ [300*84]
{
  __shared__ float acc[105];
  const int tid = threadIdx.x;
  const int k = blockIdx.x;
  if (tid < 105) acc[tid] = 0.f;
  __syncthreads();

  const float ymin = rois[k * 4 + 0], xmin = rois[k * 4 + 1];
  const float ymax = rois[k * 4 + 2], xmax = rois[k * 4 + 3];
  const float S = 1.0f / 16.0f;
  const float x1 = roundf(xmin) * S, y1 = roundf(ymin) * S;
  const float x2 = (roundf(xmax) + 1.f) * S, y2 = (roundf(ymax) + 1.f) * S;
  const float bw = fmaxf(x2 - x1, 0.1f) * (1.f / 7.f);
  const float bh = fmaxf(y2 - y1, 0.1f) * (1.f / 7.f);

  for (int u = tid; u < CCLS + CBOX; u += 256) {
    const float* maps;
    int ch, chl;
    if (u < CCLS) { maps = scoreM; ch = u;        chl = ch / 49; }
    else          { maps = bboxM;  ch = u - CCLS; chl = 21 + ch / 49; }
    int b = ch % 49;
    int i = b / 7, j = b % 7;
    float hs = fminf(fmaxf(floorf(y1 + bh * i),        0.f), (float)FH);
    float he = fminf(fmaxf(ceilf (y1 + bh * (i + 1)),  0.f), (float)FH);
    float ws = fminf(fmaxf(floorf(x1 + bw * j),        0.f), (float)FW);
    float we = fminf(fmaxf(ceilf (x1 + bw * (j + 1)),  0.f), (float)FW);
    float cnt = (he - hs) * (we - ws);
    int h0 = (int)hs, h1 = (int)he, w0 = (int)ws, w1 = (int)we;
    const float* base = maps + (size_t)ch * NP;
    float sum = 0.f;
    for (int h = h0; h < h1; ++h)
      for (int w = w0; w < w1; ++w)
        sum += base[h * FW + w];
    float pooled = (cnt > 0.f) ? sum / fmaxf(cnt, 1.f) : 0.f;
    atomicAdd(&acc[chl], pooled * (1.f / 49.f));
  }
  __syncthreads();
  if (tid < 105) {
    if (tid < 21) out[k * 21 + tid] = acc[tid];
    else          out[KROI * 21 + k * 84 + (tid - 21)] = acc[tid];
  }
}

// ---------------- launch ----------------
extern "C" void kernel_launch(void* const* d_in, const int* in_sizes, int n_in,
                              void* d_out, int out_size, void* d_ws, size_t ws_size,
                              hipStream_t stream) {
  (void)in_sizes; (void)n_in; (void)out_size; (void)ws_size;
  const float* feature  = (const float*)d_in[0];
  const float* rois     = (const float*)d_in[1];
  const float* w_double = (const float*)d_in[2];
  const float* b_double = (const float*)d_in[3];
  const float* w_score  = (const float*)d_in[4];
  const float* b_score  = (const float*)d_in[5];
  const float* w_bbox   = (const float*)d_in[6];
  const float* b_bbox   = (const float*)d_in[7];
  float* out = (float*)d_out;

  char* ws = (char*)d_ws;
  size_t off = 0;
  auto alloc = [&](size_t bytes) -> char* {
    char* p = ws + off;
    off += (bytes + 255) & ~(size_t)255;
    return p;
  };
  __bf16* featT  = (__bf16*)alloc((size_t)NP   * CIN  * 2);
  __bf16* wD     = (__bf16*)alloc((size_t)MP_D * CIN  * 2);
  __bf16* feat2T = (__bf16*)alloc((size_t)NP   * CMID * 2);
  __bf16* wS     = (__bf16*)alloc((size_t)MP_S * CMID * 2);
  __bf16* wB     = (__bf16*)alloc((size_t)MP_B * CMID * 2);
  float*  scoreM = (float*) alloc((size_t)CCLS * NP   * 4);
  float*  bboxM  = (float*) alloc((size_t)CBOX * NP   * 4);

  cvt_featT <<<(NP * CIN) / 256, 256, 0, stream>>>(feature, featT);
  cvt_w_bf16<<<(MP_D * CIN ) / 256, 256, 0, stream>>>(w_double, wD, MP_D * CIN,  MP_D * CIN);
  cvt_w_bf16<<<(MP_S * CMID) / 256, 256, 0, stream>>>(w_score,  wS, CCLS * CMID, MP_S * CMID);
  cvt_w_bf16<<<(MP_B * CMID) / 256, 256, 0, stream>>>(w_bbox,   wB, CBOX * CMID, MP_B * CMID);

  // GEMM1: feat2T[1920][1024] bf16 (transposed epilogue)
  gemm_wmma<0><<<dim3(NP / 128, MP_D / 128), 256, 0, stream>>>(wD, featT,  b_double, feat2T, MP_D, CIN,  CMID);
  // GEMM2: score maps [1029][1920] fp32
  gemm_wmma<1><<<dim3(NP / 128, MP_S / 128), 256, 0, stream>>>(wS, feat2T, b_score,  scoreM, CCLS, CMID, NP);
  // GEMM3: bbox maps [4116][1920] fp32
  gemm_wmma<1><<<dim3(NP / 128, MP_B / 128), 256, 0, stream>>>(wB, feat2T, b_bbox,   bboxM,  CBOX, CMID, NP);

  psroi_kernel<<<KROI, 256, 0, stream>>>(rois, scoreM, bboxM, out);
}
